// LSTMModel_9405978378620
// MI455X (gfx1250) — matrix operand order repaired and checked by reading, not compiled, not run on hardware
//
#include <hip/hip_runtime.h>
#include <hip/hip_bf16.h>
#include <math.h>

// Problem constants (from reference): B=16, T=4096, I=128, H=256, 4H=1024.
#define B_ 16
#define T_ 4096
#define I_ 128
#define H_ 256
#define G_ 1024   // 4*H, gate dimension

typedef __attribute__((ext_vector_type(8)))  __bf16 v8bf;
typedef __attribute__((ext_vector_type(16))) __bf16 v16bf;
typedef __attribute__((ext_vector_type(8)))  float  v8f;

// round-to-nearest-even f32 -> bf16 (as raw u16)
__device__ __forceinline__ unsigned short f2bf(float f) {
  union { float f; unsigned int u; } v; v.f = f;
  unsigned int u = v.u;
  return (unsigned short)((u + 0x7FFFu + ((u >> 16) & 1u)) >> 16);
}

// Build a 16x32 bf16 A-fragment for one lane from row-major LDS.
// ISA layout (16-bit A 16x32): lanes 0-15: M=lane, K = {0..7, 16..23};
// lanes 16-31: M=lane-16, K = {8..15, 24..31}. Two contiguous 16B chunks.
__device__ __forceinline__ v16bf lds_a_frag(const unsigned short* p) {
  union { v16bf v; v8bf h[2]; } u;
  u.h[0] = *(const v8bf*)(p);        // K = koff .. koff+7
  u.h[1] = *(const v8bf*)(p + 16);   // K = koff+16 .. koff+23
  return u.v;
}

// B-fragment: pre-packed contiguously per lane (32 bytes).
__device__ __forceinline__ v16bf ld_b_frag(const unsigned short* p) {
  return *(const v16bf*)p;
}

// ---------------------------------------------------------------------------
// Pack W [G x K] fp32 (row-major, gate-major) into WMMA B-fragment order:
// packed[((nt*KT + kt)*32 + lane)*16 + e] = bf16( W[(nt*16 + lane%16) * K + kt*32 + k(lane,e)] )
// ISA B layout (32x16 bf16): lane l holds column N=l%16; element e -> K = 8*(l/16) + (e&7) + 16*(e>>3).
// ---------------------------------------------------------------------------
__global__ void pack_w_kernel(const float* __restrict__ W,
                              unsigned short* __restrict__ o, int G, int K) {
  int idx = blockIdx.x * blockDim.x + threadIdx.x;
  if (idx >= G * K) return;
  int KT   = K >> 5;
  int e    = idx & 15;
  int lane = (idx >> 4) & 31;
  int kt   = (idx >> 9) % KT;
  int nt   = idx / (KT << 9);
  int n = nt * 16 + (lane & 15);
  int k = kt * 32 + ((lane >> 4) << 3) + (e & 7) + ((e >> 3) << 4);
  o[idx] = f2bf(W[n * K + k]);
}

// ---------------------------------------------------------------------------
// xW[t][b][g] = sum_k in[..][k] * W[g][k] + bias[g]   (fp32 out, bf16 WMMA)
// One block = one timestep (M=16 = all batches) x 256 gate columns.
// grid = (T, G/256), block = 256 threads (8 waves x 2 N-tiles).
// ---------------------------------------------------------------------------
__global__ void gemm_xw_kernel(const float* __restrict__ in,
                               long long t_stride, long long b_stride,
                               const unsigned short* __restrict__ Wp,
                               const float* __restrict__ bias,
                               float* __restrict__ xW, int K) {
  extern __shared__ unsigned short lds_a[];   // [16][K+8] bf16
  const int KP  = K + 8;
  const int t   = blockIdx.x;
  const int tid = threadIdx.x;

  // stage A (input rows for this t) into LDS as bf16
  for (int i = tid; i < 16 * K; i += 256) {
    int m = i / K, k = i - m * K;
    lds_a[m * KP + k] =
        f2bf(in[(long long)t * t_stride + (long long)m * b_stride + k]);
  }
  __syncthreads();

  const int wave = tid >> 5, lane = tid & 31;
  const int n0   = blockIdx.y * 256 + wave * 32;
  const int KT   = K >> 5;
  const int mrow = lane & 15;
  const int koff = (lane >> 4) * 8;
  const unsigned short* arow = lds_a + mrow * KP;

  v8f acc0 = {}; v8f acc1 = {};
  for (int kf = 0; kf < KT; ++kf) {
    v16bf a  = lds_a_frag(arow + kf * 32 + koff);
    v16bf b0 = ld_b_frag(Wp + (((((n0 >> 4)    ) * KT + kf) * 32 + lane) << 4));
    v16bf b1 = ld_b_frag(Wp + (((((n0 >> 4) + 1) * KT + kf) * 32 + lane) << 4));
    acc0 = __builtin_amdgcn_wmma_f32_16x16x32_bf16(false, a, false, b0,
                                                   (short)0, acc0, false, false);
    acc1 = __builtin_amdgcn_wmma_f32_16x16x32_bf16(false, a, false, b1,
                                                   (short)0, acc1, false, false);
  }

  // D layout: vgpr r, lane l -> M = r + 8*(l>>4), N = l&15
  const int nlo = lane & 15;
  const int mh  = (lane >> 4) * 8;
  float bia0 = bias[n0 + nlo], bia1 = bias[n0 + 16 + nlo];
#pragma unroll
  for (int r = 0; r < 8; ++r) {
    long long row = ((long long)t * 16 + (mh + r)) * G_;
    xW[row + n0 + nlo]      = acc0[r] + bia0;
    xW[row + n0 + 16 + nlo] = acc1[r] + bia1;
  }
}

// ---------------------------------------------------------------------------
// Persistent LSTM scan for one layer. Single workgroup (32 wave32 waves) owns
// the whole recurrence:
//   - Whh bf16 B-fragments live in VGPRs for all 4096 steps (persistent RNN)
//   - h is exchanged via LDS; c lives in per-thread registers
//   - xW[t] slices are double-buffered in LDS via GLOBAL_LOAD_ASYNC_TO_LDS_B128,
//     DMA of step t+1 overlaps compute of step t (ASYNCcnt / s_wait_asynccnt)
//   wave w -> gate columns [32w, 32w+32); gate type = w/8 (i,f,g,o).
//   thread -> 4 fixed (b, j) cell states.
// LDS: [xw ping 64KB][xw pong 64KB][gates 66KB][h bf16 8.4KB] ~ 203KB (<320KB WGP)
// ---------------------------------------------------------------------------
#define XWB_BYTES (16 * G_ * 4)          // 65536 per buffer
#define GATES_OFF (2 * XWB_BYTES)        // 131072
#define GS_ (G_ + 8)                     // padded gate row stride (floats)
#define HS_ (H_ + 8)                     // padded h row stride (bf16)
#define HBF_OFF (GATES_OFF + 16 * GS_ * 4)

__global__ void lstm_scan_kernel(const float* __restrict__ xW,           // [T][16][1024] fp32 (bias folded)
                                 const unsigned short* __restrict__ WhhP,// packed bf16
                                 float* __restrict__ hout,               // layer out
                                 long long ho_t, long long ho_b) {
  extern __shared__ char smem[];
  float* xwlds        = (float*)smem;                         // 2 x [16*1024] f32
  float* gates        = (float*)(smem + GATES_OFF);           // 16*GS f32
  unsigned short* hbf = (unsigned short*)(smem + HBF_OFF);    // 16*HS bf16

  const int tid  = threadIdx.x;
  const int wave = tid >> 5, lane = tid & 31;
  const int n0   = wave * 32;
  const int gt   = wave >> 3;          // 0:i 1:f 2:g 3:o  (torch order)
  const int KT   = H_ >> 5;            // 8
  const int mrow = lane & 15;
  const int koff = (lane >> 4) * 8;
  const int nlo  = lane & 15;
  const int mh   = (lane >> 4) * 8;

  // zero h_{-1}
  for (int i = tid; i < 16 * HS_; i += 1024) hbf[i] = 0;

  // persistent weights: 2 N-tiles x 8 K-fragments x 8 VGPRs = 128 VGPRs/wave
  v16bf Bf0[8], Bf1[8];
#pragma unroll
  for (int kf = 0; kf < 8; ++kf) {
    Bf0[kf] = ld_b_frag(WhhP + (((((n0 >> 4)    ) * KT + kf) * 32 + lane) << 4));
    Bf1[kf] = ld_b_frag(WhhP + (((((n0 >> 4) + 1) * KT + kf) * 32 + lane) << 4));
  }

  // per-thread cell state: batch b, hidden j0..j0+3
  const int b  = (tid * 4) >> 8;
  const int j0 = (tid * 4) & 255;
  float c0 = 0.f, c1 = 0.f, c2 = 0.f, c3 = 0.f;

  // prime the pipeline: async-DMA xW[0] into LDS buffer 0
  // (dynamic LDS starts at offset 0: no static __shared__ in this kernel)
  {
#pragma unroll
    for (int ii = 0; ii < 4; ++ii) {
      unsigned q = (unsigned)tid + 1024u * ii;          // 16B chunk index
      unsigned lds_off = q * 16u;
      unsigned long long ga = (unsigned long long)(const void*)(xW + q * 4);
      asm volatile("global_load_async_to_lds_b128 %0, %1, off"
                   :: "v"(lds_off), "v"(ga) : "memory");
    }
    asm volatile("s_wait_asynccnt 0x0" ::: "memory");
  }
  __syncthreads();

  const unsigned short* arow = hbf + mrow * HS_;

  for (int t = 0; t < T_; ++t) {
    // (a) kick off async DMA of xW[t+1] into the other LDS buffer
    if (t + 1 < T_) {
      const float* src = xW + ((long long)(t + 1) * 16) * G_;
      unsigned dstbase = (unsigned)(((t + 1) & 1) * XWB_BYTES);
#pragma unroll
      for (int ii = 0; ii < 4; ++ii) {
        unsigned q = (unsigned)tid + 1024u * ii;
        unsigned lds_off = dstbase + q * 16u;
        unsigned long long ga = (unsigned long long)(const void*)(src + q * 4);
        asm volatile("global_load_async_to_lds_b128 %0, %1, off"
                     :: "v"(lds_off), "v"(ga) : "memory");
      }
    }

    // (b) gates_pre = h_{t-1} @ Whh^T  (bf16 WMMA, f32 acc, weights in VGPRs)
    v8f acc0 = {}; v8f acc1 = {};
#pragma unroll
    for (int kf = 0; kf < 8; ++kf) {
      v16bf a = lds_a_frag(arow + kf * 32 + koff);
      acc0 = __builtin_amdgcn_wmma_f32_16x16x32_bf16(false, a, false, Bf0[kf],
                                                     (short)0, acc0, false, false);
      acc1 = __builtin_amdgcn_wmma_f32_16x16x32_bf16(false, a, false, Bf1[kf],
                                                     (short)0, acc1, false, false);
    }

    // (c) += xW[t] from LDS ping-pong buffer (bias folded), activate, publish
    const float* xrow = xwlds + (t & 1) * (16 * G_);
#pragma unroll
    for (int r = 0; r < 8; ++r) {
      int m = mh + r;
      float g0 = acc0[r] + xrow[m * G_ + n0 + nlo];
      float g1 = acc1[r] + xrow[m * G_ + n0 + 16 + nlo];
      if (gt == 2) { g0 = tanhf(g0); g1 = tanhf(g1); }
      else         { g0 = 1.f / (1.f + __expf(-g0));
                     g1 = 1.f / (1.f + __expf(-g1)); }
      gates[m * GS_ + n0 + nlo]      = g0;
      gates[m * GS_ + n0 + 16 + nlo] = g1;
    }
    __syncthreads();   // (d)

    // (e) elementwise cell/hidden update; write h to global and (bf16) to LDS
    {
      const float* gb = gates + b * GS_;
      float iv, fv, gv, ov, cc;
      float hv0, hv1, hv2, hv3;
      iv = gb[j0 + 0]; fv = gb[j0 + 256]; gv = gb[j0 + 512]; ov = gb[j0 + 768];
      cc = fv * c0 + iv * gv; c0 = cc; hv0 = ov * tanhf(cc);
      iv = gb[j0 + 1]; fv = gb[j0 + 257]; gv = gb[j0 + 513]; ov = gb[j0 + 769];
      cc = fv * c1 + iv * gv; c1 = cc; hv1 = ov * tanhf(cc);
      iv = gb[j0 + 2]; fv = gb[j0 + 258]; gv = gb[j0 + 514]; ov = gb[j0 + 770];
      cc = fv * c2 + iv * gv; c2 = cc; hv2 = ov * tanhf(cc);
      iv = gb[j0 + 3]; fv = gb[j0 + 259]; gv = gb[j0 + 515]; ov = gb[j0 + 771];
      cc = fv * c3 + iv * gv; c3 = cc; hv3 = ov * tanhf(cc);

      float* op = hout + (long long)t * ho_t + (long long)b * ho_b + j0;
      op[0] = hv0; op[1] = hv1; op[2] = hv2; op[3] = hv3;
      unsigned short* hb = hbf + b * HS_ + j0;
      hb[0] = f2bf(hv0); hb[1] = f2bf(hv1); hb[2] = f2bf(hv2); hb[3] = f2bf(hv3);
    }

    // (f) drain this thread's async DMA, then (g) make it LDS-visible to all
    asm volatile("s_wait_asynccnt 0x0" ::: "memory");
    __syncthreads();
  }
}

// ---------------------------------------------------------------------------
// Launch: pack weights -> GEMM0 -> scan0 -> GEMM1 -> scan1
// ws layout: [xW fp32 256MB][h1 fp32 64MB][packed bf16 weights ~1.75MB]
// ---------------------------------------------------------------------------
extern "C" void kernel_launch(void* const* d_in, const int* in_sizes, int n_in,
                              void* d_out, int out_size, void* d_ws, size_t ws_size,
                              hipStream_t stream) {
  const float* x    = (const float*)d_in[0];
  const float* Wih0 = (const float*)d_in[1];  // [1024,128]
  const float* Whh0 = (const float*)d_in[2];  // [1024,256]
  const float* b0   = (const float*)d_in[3];  // [1024]
  const float* Wih1 = (const float*)d_in[4];  // [1024,256]
  const float* Whh1 = (const float*)d_in[5];  // [1024,256]
  const float* b1   = (const float*)d_in[6];  // [1024]
  float* out = (float*)d_out;                 // [16,4096,256]

  char* ws = (char*)d_ws;
  float* xWbuf = (float*)ws;                                        // T*16*1024 f32
  float* h1    = (float*)(ws + (size_t)T_ * 16 * G_ * 4);           // T*16*256 f32
  unsigned short* wp =
      (unsigned short*)(ws + (size_t)T_ * 16 * G_ * 4 + (size_t)T_ * 16 * H_ * 4);
  unsigned short* Wih0p = wp;
  unsigned short* Whh0p = Wih0p + (size_t)G_ * I_;
  unsigned short* Wih1p = Whh0p + (size_t)G_ * H_;
  unsigned short* Whh1p = Wih1p + (size_t)G_ * H_;

  pack_w_kernel<<<(G_ * I_ + 255) / 256, 256, 0, stream>>>(Wih0, Wih0p, G_, I_);
  pack_w_kernel<<<(G_ * H_ + 255) / 256, 256, 0, stream>>>(Whh0, Whh0p, G_, H_);
  pack_w_kernel<<<(G_ * H_ + 255) / 256, 256, 0, stream>>>(Wih1, Wih1p, G_, H_);
  pack_w_kernel<<<(G_ * H_ + 255) / 256, 256, 0, stream>>>(Whh1, Whh1p, G_, H_);

  size_t scan_lds = (size_t)HBF_OFF + (size_t)16 * HS_ * 2;  // ~203KB

  // layer 0: x is [B,T,I] -> row b at t: offset t*I + b*(T*I)
  {
    dim3 g(T_, G_ / 256);
    gemm_xw_kernel<<<g, 256, (size_t)16 * (I_ + 8) * 2, stream>>>(
        x, (long long)I_, (long long)T_ * I_, Wih0p, b0, xWbuf, I_);
  }
  // h1 stored [t][b][h]
  lstm_scan_kernel<<<1, 1024, scan_lds, stream>>>(
      xWbuf, Whh0p, h1, (long long)16 * H_, (long long)H_);

  // layer 1: input h1 [t][b][h]
  {
    dim3 g(T_, G_ / 256);
    gemm_xw_kernel<<<g, 256, (size_t)16 * (H_ + 8) * 2, stream>>>(
        h1, (long long)16 * H_, (long long)H_, Wih1p, b1, xWbuf, H_);
  }
  // output d_out is [b][t][h]
  lstm_scan_kernel<<<1, 1024, scan_lds, stream>>>(
      xWbuf, Whh1p, out, (long long)H_, (long long)T_ * H_);
}
